// LogLinearMamba2_13752485282228
// MI455X (gfx1250) — compile-verified, matrix operand order checked by reading, not run on hardware
//
#include <hip/hip_runtime.h>
#include <hip/hip_bf16.h>

// ---------------- model constants ----------------
#define T        1024
#define HID      1024
#define NH       32
#define HP       64
#define NS       128
#define NLV      15
#define KW       4
#define INTER    2048
#define CONV_DIM 2304
#define PROJ     4864
#define DT_OFF   (INTER + CONV_DIM)   // 4352
#define DL_OFF   (DT_OFF + NH)        // 4384
#define XB_OFF   2048                 // B offset inside conv channels
#define XC_OFF   2176                 // C offset inside conv channels

typedef __attribute__((ext_vector_type(16))) __bf16 v16bf;
typedef __attribute__((ext_vector_type(8)))  __bf16 v8bf;
typedef __attribute__((ext_vector_type(8)))  float  v8f;

// ---------------- helpers ----------------
__device__ __forceinline__ __bf16 f2bf(float f) {
    unsigned u = __builtin_bit_cast(unsigned, f);
    u += 0x7FFFu + ((u >> 16) & 1u);           // round to nearest even
    unsigned short h = (unsigned short)(u >> 16);
    return __builtin_bit_cast(__bf16, h);
}
__device__ __forceinline__ float softplus_f(float x) {
    return (x > 20.f) ? x : log1pf(__expf(x));
}
__device__ __forceinline__ float silu_f(float x) {
    return x / (1.f + __expf(-x));
}

// Load a 16x32 bf16 WMMA fragment (A-layout, ISA §7.12.2) from a row-major
// matrix: lane L -> row (r0 + L%16); K chunks (L/16)*8+[0..7] and +16.
// Both 8-element chunks are 16B-aligned contiguous loads.
__device__ __forceinline__ v16bf ldfrag(const __bf16* base, int ld, int r0, int k0, int lane) {
    const int m  = lane & 15;
    const int kh = lane >> 4;
    const __bf16* p = base + (size_t)(r0 + m) * ld + k0 + kh * 8;
    v8bf lo = *(const v8bf*)p;
    v8bf hi = *(const v8bf*)(p + 16);
    v16bf f;
#pragma unroll
    for (int i = 0; i < 8; ++i) { f[i] = lo[i]; f[8 + i] = hi[i]; }
    return f;
}

#define WMMA_BF16(A_, B_, C_) \
    __builtin_amdgcn_wmma_f32_16x16x32_bf16(false, (A_), false, (B_), (short)0, (C_), false, false)

// ---------------- kernels ----------------

// f32 -> bf16 pack
__global__ void k_cvt_bf16(__bf16* __restrict__ dst, const float* __restrict__ src, int n) {
    int i = blockIdx.x * blockDim.x + threadIdx.x;
    if (i < n) dst[i] = f2bf(src[i]);
}

// C[M,N] = A[M,K](bf16) * W[N,K]^T(bf16) + bias[N], f32 out.
// One wave per 16x64 tile. All five fragments of a K-step are loaded as
// simultaneously-live values BEFORE the 4-WMMA burst so the scheduler can
// issue the whole load clause early and use partial s_wait_loadcnt (instead
// of a full drain before every WMMA), overlapping next-step loads with math.
__global__ void __launch_bounds__(256)
k_gemm_bf16(const __bf16* __restrict__ A, const __bf16* __restrict__ W,
            const float* __restrict__ bias, float* __restrict__ C,
            int M, int N, int K) {
    const int lane = threadIdx.x & 31;
    const int wg   = (blockIdx.x * blockDim.x + threadIdx.x) >> 5;  // global wave id
    const int tilesN = N >> 6;                    // 64-wide N tiles
    const int tm = wg / tilesN, tn = wg % tilesN;
    if (tm * 16 >= M) return;                     // wave-uniform

    v8f acc0, acc1, acc2, acc3;
    {
        const float b0 = bias[tn * 64 +  0 + (lane & 15)];
        const float b1 = bias[tn * 64 + 16 + (lane & 15)];
        const float b2 = bias[tn * 64 + 32 + (lane & 15)];
        const float b3 = bias[tn * 64 + 48 + (lane & 15)];
#pragma unroll
        for (int r = 0; r < 8; ++r) { acc0[r] = b0; acc1[r] = b1; acc2[r] = b2; acc3[r] = b3; }
    }

    const __bf16* arow = A + (size_t)(tm * 16 + (lane & 15)) * K;
    for (int k0 = 0; k0 < K; k0 += 32) {
        v16bf a  = ldfrag(A, K, tm * 16, k0, lane);
        v16bf b0 = ldfrag(W, K, tn * 64 +  0, k0, lane);
        v16bf b1 = ldfrag(W, K, tn * 64 + 16, k0, lane);
        v16bf b2 = ldfrag(W, K, tn * 64 + 32, k0, lane);
        v16bf b3 = ldfrag(W, K, tn * 64 + 48, k0, lane);
        __builtin_prefetch((const void*)(arow + k0 + 128), 0, 1);   // global_prefetch_b8
        acc0 = WMMA_BF16(a, b0, acc0);
        acc1 = WMMA_BF16(a, b1, acc1);
        acc2 = WMMA_BF16(a, b2, acc2);
        acc3 = WMMA_BF16(a, b3, acc3);
    }
    const int rbase = tm * 16 + 8 * (lane >> 4);
    const int ncol  = tn * 64 + (lane & 15);
#pragma unroll
    for (int r = 0; r < 8; ++r) {
        float* crow = C + (size_t)(rbase + r) * N + ncol;
        crow[0]  = acc0[r];
        crow[16] = acc1[r];
        crow[32] = acc2[r];
        crow[48] = acc3[r];
    }
}

// depthwise causal conv1d (width 4) + SiLU over the xBC slice of zx
__global__ void k_conv(const float* __restrict__ zx, const float* __restrict__ cw,
                       float* __restrict__ out) {
    int i = blockIdx.x * blockDim.x + threadIdx.x;
    if (i >= T * CONV_DIM) return;
    const int t = i / CONV_DIM, c = i % CONV_DIM;
    float acc = 0.f;
#pragma unroll
    for (int w = 0; w < KW; ++w) {
        int s = t + w - (KW - 1);
        if (s >= 0) acc += zx[(size_t)s * PROJ + INTER + c] * cw[c * KW + w];
    }
    out[i] = silu_f(acc);
}

// per-(t,h): dt_sp = softplus(dt+bias); g[h][t] = -exp(A_log)*dt_sp;
// Ls[h][t][l] = softplus(L_param*dl), padded to 16 levels.
__global__ void k_prep(const float* __restrict__ zx, const float* __restrict__ dt_bias,
                       const float* __restrict__ A_log, const float* __restrict__ L_param,
                       float* __restrict__ dt_sp, float* __restrict__ g,
                       float* __restrict__ Ls) {
    int i = blockIdx.x * blockDim.x + threadIdx.x;
    if (i >= T * NH) return;
    const int t = i / NH, h = i % NH;
    const float d = softplus_f(zx[(size_t)t * PROJ + DT_OFF + h] + dt_bias[h]);
    dt_sp[t * NH + h] = d;
    g[h * T + t] = -__expf(A_log[h]) * d;
    float* lsrow = Ls + ((size_t)h * T + t) * 16;
#pragma unroll
    for (int l = 0; l < NLV; ++l)
        lsrow[l] = softplus_f(L_param[h * NLV + l] * zx[(size_t)t * PROJ + DL_OFF + h * NLV + l]);
    lsrow[NLV] = 0.f;
}

// serial per-head cumulative log-decay (32 independent 1024-length scans)
__global__ void k_cumsum(const float* __restrict__ g, float* __restrict__ cg) {
    int h = threadIdx.x;
    if (h >= NH) return;
    float acc = 0.f;
    for (int t = 0; t < T; ++t) { acc += g[h * T + t]; cg[h * T + t] = acc; }
}

// vT[h][p][t] = bf16(x[t][h*P+p] * dt_sp[t][h])  (transposed so PV B-fragments
// are contiguous-in-s loads)
__global__ void k_pack_v(const float* __restrict__ conv, const float* __restrict__ dt_sp,
                         __bf16* __restrict__ vT) {
    int i = blockIdx.x * blockDim.x + threadIdx.x;
    if (i >= NH * HP * T) return;
    const int t = i % T, hp = i / T;
    const int h = hp >> 6;
    vT[i] = f2bf(conv[(size_t)t * CONV_DIM + hp] * dt_sp[t * NH + h]);
}

// bf16 copies of B (keys) and C (queries)
__global__ void k_pack_bc(const float* __restrict__ conv, __bf16* __restrict__ Bb,
                          __bf16* __restrict__ Cb) {
    int i = blockIdx.x * blockDim.x + threadIdx.x;
    if (i >= T * 2 * NS) return;
    const int t = i / (2 * NS), c = i % (2 * NS);
    if (c < NS) Bb[t * NS + c] = f2bf(conv[(size_t)t * CONV_DIM + XB_OFF + c]);
    else        Cb[t * NS + (c - NS)] = f2bf(conv[(size_t)t * CONV_DIM + XC_OFF + (c - NS)]);
}

// Fused triangular attention: per wave one (head, 16-query tile).
// Query fragments and cg[t] are s-invariant registers. Per s-block:
// S = C·Bᵀ (4 chained WMMAs over N=128, loads hoisted ahead of the chain),
// gate with exp(cg_t-cg_s)·Ls[level], mask, stage 16x32 bf16 P tile in LDS,
// then Y += P·V (4 independent WMMAs, vT fragments preloaded).
__global__ void __launch_bounds__(256)
k_attn(const __bf16* __restrict__ Cb, const __bf16* __restrict__ Bb,
       const __bf16* __restrict__ vT, const float* __restrict__ cg,
       const float* __restrict__ Ls, const int* __restrict__ lvl,
       float* __restrict__ yatt) {
    __shared__ __align__(16) __bf16 pbuf[8][16 * 32];
    const int wave = threadIdx.x >> 5, lane = threadIdx.x & 31;
    const int h  = blockIdx.x >> 3;            // 8 q-blocks of 128 per head
    const int qb = blockIdx.x & 7;
    const int q0 = qb * 128 + wave * 16;
    const float* cgh = cg + (size_t)h * T;
    const float* lsh = Ls + (size_t)h * T * 16;
    const __bf16* vh = vT + (size_t)h * HP * T;
    __bf16* pw = &pbuf[wave][0];

    // s-invariant query fragments (4 x 16x32 over the N=128 contraction)
    v16bf cfr0 = ldfrag(Cb, NS, q0,  0, lane);
    v16bf cfr1 = ldfrag(Cb, NS, q0, 32, lane);
    v16bf cfr2 = ldfrag(Cb, NS, q0, 64, lane);
    v16bf cfr3 = ldfrag(Cb, NS, q0, 96, lane);
    // s-invariant per-row cumulative decays
    float cgt[8];
#pragma unroll
    for (int r = 0; r < 8; ++r) cgt[r] = cgh[q0 + r + 8 * (lane >> 4)];

    v8f yacc0, yacc1, yacc2, yacc3;
#pragma unroll
    for (int r = 0; r < 8; ++r) { yacc0[r] = 0.f; yacc1[r] = 0.f; yacc2[r] = 0.f; yacc3[r] = 0.f; }

    const int sblocks = (q0 + 16 + 31) >> 5;   // cover s in [0, q0+16)
    for (int sb = 0; sb < sblocks; ++sb) {
        const int s0 = sb * 32;
#pragma unroll
        for (int half = 0; half < 2; ++half) {
            const int ss = s0 + half * 16;
            v8f sc;
#pragma unroll
            for (int r = 0; r < 8; ++r) sc[r] = 0.f;
            if (ss < q0 + 16) {                // wave-uniform: skip fully-masked subtiles
                v16bf b0 = ldfrag(Bb, NS, ss,  0, lane);
                v16bf b1 = ldfrag(Bb, NS, ss, 32, lane);
                v16bf b2 = ldfrag(Bb, NS, ss, 64, lane);
                v16bf b3 = ldfrag(Bb, NS, ss, 96, lane);
                sc = WMMA_BF16(cfr0, b0, sc);
                sc = WMMA_BF16(cfr1, b1, sc);
                sc = WMMA_BF16(cfr2, b2, sc);
                sc = WMMA_BF16(cfr3, b3, sc);
                const int scol = ss + (lane & 15);
                const float cgs = cgh[scol];
#pragma unroll
                for (int r = 0; r < 8; ++r) {
                    const int trow = q0 + r + 8 * (lane >> 4);
                    float v = 0.f;
                    if (scol <= trow) {
                        const int lv = lvl[(size_t)trow * T + scol];
                        v = sc[r] * __expf(cgt[r] - cgs) * lsh[trow * 16 + lv];
                    }
                    sc[r] = v;
                }
            }
            // stage C/D-layout tile into LDS as row-major 16x32 bf16
            const int n = lane & 15;
#pragma unroll
            for (int r = 0; r < 8; ++r) {
                const int m = r + 8 * (lane >> 4);
                pw[m * 32 + half * 16 + n] = f2bf(sc[r]);
            }
        }
        // P(16x32) @ V(32x64): A-frag from LDS (same-wave, compiler inserts
        // dscnt wait); all four vT fragments preloaded for partial loadcnt waits.
        v16bf vb0 = ldfrag(vh, T,  0, s0, lane);
        v16bf vb1 = ldfrag(vh, T, 16, s0, lane);
        v16bf vb2 = ldfrag(vh, T, 32, s0, lane);
        v16bf vb3 = ldfrag(vh, T, 48, s0, lane);
        v16bf pa  = ldfrag(pw, 32, 0, 0, lane);
        yacc0 = WMMA_BF16(pa, vb0, yacc0);
        yacc1 = WMMA_BF16(pa, vb1, yacc1);
        yacc2 = WMMA_BF16(pa, vb2, yacc2);
        yacc3 = WMMA_BF16(pa, vb3, yacc3);
    }
    const int n = lane & 15;
#pragma unroll
    for (int r = 0; r < 8; ++r) {
        const int t = q0 + r + 8 * (lane >> 4);
        float* yrow = yatt + (size_t)t * INTER + h * HP + n;
        yrow[0]  = yacc0[r];
        yrow[16] = yacc1[r];
        yrow[32] = yacc2[r];
        yrow[48] = yacc3[r];
    }
}

// y_tot = y_att + x*D;  yg = y_tot*silu(z);  RMSNorm(yg)*w -> bf16
__global__ void __launch_bounds__(256)
k_gatenorm(const float* __restrict__ yatt, const float* __restrict__ conv,
           const float* __restrict__ Dp, const float* __restrict__ zx,
           const float* __restrict__ rw, __bf16* __restrict__ yn) {
    __shared__ float ygbuf[INTER];
    __shared__ float red[256];
    const int t = blockIdx.x;
    float part = 0.f;
    for (int c = threadIdx.x; c < INTER; c += 256) {
        const int h = c >> 6;
        const float yt = yatt[(size_t)t * INTER + c] + conv[(size_t)t * CONV_DIM + c] * Dp[h];
        const float z  = zx[(size_t)t * PROJ + c];
        const float yg = yt * silu_f(z);
        ygbuf[c] = yg;
        part += yg * yg;
    }
    red[threadIdx.x] = part;
    __syncthreads();
    for (int s = 128; s > 0; s >>= 1) {
        if (threadIdx.x < s) red[threadIdx.x] += red[threadIdx.x + s];
        __syncthreads();
    }
    const float r = rsqrtf(red[0] / (float)INTER + 1e-5f);
    for (int c = threadIdx.x; c < INTER; c += 256)
        yn[(size_t)t * INTER + c] = f2bf(ygbuf[c] * r * rw[c]);
}

// ---------------- launcher ----------------
extern "C" void kernel_launch(void* const* d_in, const int* in_sizes, int n_in,
                              void* d_out, int out_size, void* d_ws, size_t ws_size,
                              hipStream_t stream) {
    const float* hs      = (const float*)d_in[0];
    const float* w_in    = (const float*)d_in[1];
    const float* b_in    = (const float*)d_in[2];
    const float* conv_w  = (const float*)d_in[3];
    const float* dt_bias = (const float*)d_in[4];
    const float* A_log   = (const float*)d_in[5];
    const float* L_param = (const float*)d_in[6];
    const float* Dp      = (const float*)d_in[7];
    const float* rw      = (const float*)d_in[8];
    const float* w_out   = (const float*)d_in[9];
    const float* b_out   = (const float*)d_in[10];
    const int*   lvl     = (const int*)d_in[11];
    float* out = (float*)d_out;

    char* ws = (char*)d_ws;
    size_t off = 0;
    auto alloc = [&](size_t bytes) -> char* {
        char* p = ws + off;
        off += (bytes + 255) & ~(size_t)255;
        return p;
    };
    __bf16* hs_bf   = (__bf16*)alloc((size_t)T * HID * 2);
    __bf16* win_bf  = (__bf16*)alloc((size_t)PROJ * HID * 2);
    __bf16* wout_bf = (__bf16*)alloc((size_t)HID * INTER * 2);
    float*  zx      = (float*)alloc((size_t)T * PROJ * 4);
    float*  convout = (float*)alloc((size_t)T * CONV_DIM * 4);
    float*  dt_sp   = (float*)alloc((size_t)T * NH * 4);
    float*  g       = (float*)alloc((size_t)NH * T * 4);
    float*  cgbuf   = (float*)alloc((size_t)NH * T * 4);
    float*  Ls      = (float*)alloc((size_t)NH * T * 16 * 4);
    __bf16* vT      = (__bf16*)alloc((size_t)NH * HP * T * 2);
    __bf16* Bb      = (__bf16*)alloc((size_t)T * NS * 2);
    __bf16* Cb      = (__bf16*)alloc((size_t)T * NS * 2);
    float*  yatt    = (float*)alloc((size_t)T * INTER * 4);
    __bf16* yn      = (__bf16*)alloc((size_t)T * INTER * 2);

    // 1) pack bf16 operands
    k_cvt_bf16<<<(T * HID + 255) / 256, 256, 0, stream>>>(hs_bf, hs, T * HID);
    k_cvt_bf16<<<(PROJ * HID + 255) / 256, 256, 0, stream>>>(win_bf, w_in, PROJ * HID);
    k_cvt_bf16<<<(HID * INTER + 255) / 256, 256, 0, stream>>>(wout_bf, w_out, HID * INTER);

    // 2) in_proj GEMM: (T x HID) * (PROJ x HID)^T -> (T x PROJ); 16x64 per wave
    k_gemm_bf16<<<(T / 16) * (PROJ / 64) / 8, 256, 0, stream>>>(hs_bf, win_bf, b_in, zx,
                                                                T, PROJ, HID);
    // 3) conv + SiLU
    k_conv<<<(T * CONV_DIM + 255) / 256, 256, 0, stream>>>(zx, conv_w, convout);
    // 4) dt/g/Ls
    k_prep<<<(T * NH + 255) / 256, 256, 0, stream>>>(zx, dt_bias, A_log, L_param, dt_sp, g, Ls);
    // 5) per-head cumulative log-decay
    k_cumsum<<<1, 32, 0, stream>>>(g, cgbuf);
    // 6) pack v (transposed) and B/C bf16
    k_pack_v<<<(NH * HP * T + 255) / 256, 256, 0, stream>>>(convout, dt_sp, vT);
    k_pack_bc<<<(T * 2 * NS + 255) / 256, 256, 0, stream>>>(convout, Bb, Cb);
    // 7) fused decayed hierarchical attention
    k_attn<<<NH * (T / 128), 256, 0, stream>>>(Cb, Bb, vT, cgbuf, Ls, lvl, yatt);
    // 8) D residual + gated RMSNorm -> bf16 activations
    k_gatenorm<<<T, 256, 0, stream>>>(yatt, convout, Dp, zx, rw, yn);
    // 9) out_proj GEMM: (T x INTER) * (HID x INTER)^T -> (T x HID); 16x64 per wave
    k_gemm_bf16<<<(T / 16) * (HID / 64) / 8, 256, 0, stream>>>(yn, wout_bf, b_out, out,
                                                               T, HID, INTER);
}
